// UniMolModel_27599459844775
// MI455X (gfx1250) — matrix operand order repaired and checked by reading, not compiled, not run on hardware
//
#include <hip/hip_runtime.h>
#include <math.h>

typedef __bf16 bf16;
typedef __attribute__((ext_vector_type(16))) __bf16 v16bf;
typedef __attribute__((ext_vector_type(8)))  __bf16 v8bf;
typedef __attribute__((ext_vector_type(8)))  float  v8f;

#define Bb 4
#define Ll 256
#define Dd 512
#define Hh 64
#define FFNf 2048
#define Vv 32
#define Kk 128
#define NLAYERS 8
#define BL   (Bb*Ll)        // 1024
#define BLL  (Bb*Ll*Ll)     // 262144
#define BHLL (Bb*Hh*Ll*Ll)  // 16777216

// ---------------------------------------------------------------- helpers ---
__device__ __forceinline__ float gelu_f(float x) {
  return 0.5f * x * (1.0f + erff(x * 0.70710678118654752f));
}
__device__ __forceinline__ v8f vzero8() {
  v8f z = {0.f,0.f,0.f,0.f,0.f,0.f,0.f,0.f};
  return z;
}
__device__ __forceinline__ v8f wmma_bf16(v16bf a, v16bf b, v8f c) {
  // D = A(16x32 bf16) * B(32x16 bf16) + C(16x16 f32)
  return __builtin_amdgcn_wmma_f32_16x16x32_bf16(false, a, false, b, (short)0, c, false, false);
}

// A fragment (16x32 bf16) from row-major tile (stride ld), rows m..m+15, k offset k0.
// Per-lane data is one contiguous 16-element run [k0+16h .. +15] (matches B fragment)
// -> two 16B vector loads.
__device__ __forceinline__ v16bf load_a_frag(const bf16* tile, int ld, int m, int k0) {
  int lane = threadIdx.x & 31;
  const bf16* row = tile + (m + (lane & 15)) * ld + k0 + ((lane >> 4) << 4);
  v8bf lo = *(const v8bf*)row;
  v8bf hi = *(const v8bf*)(row + 8);
  return __builtin_shufflevector(lo, hi, 0,1,2,3,4,5,6,7,8,9,10,11,12,13,14,15);
}

// B fragment (32x16 bf16) from K-major B^T[N][K] (row stride ldk): per lane one
// contiguous 32B run at [n, k0+16*half .. +15]  -> single v16bf load.
__device__ __forceinline__ v16bf load_bt_frag(const bf16* bt, int ldk, int n0, int k0) {
  int lane = threadIdx.x & 31;
  return *(const v16bf*)(bt + (size_t)(n0 + (lane & 15)) * ldk + k0 + ((lane >> 4) << 4));
}

// ------------------------------------------------------------ conversions ---
// src [batch, rows, cols] f32 -> dst [batch, cols, rows] bf16 (per-batch transpose)
__global__ void cvt_t_kernel(const float* __restrict__ src, bf16* __restrict__ dst,
                             int rows, int cols) {
  int bt = blockIdx.y;
  size_t nm = (size_t)rows * cols;
  int i = blockIdx.x * 256 + threadIdx.x;
  if (i < (int)nm) {
    int r = i / cols, c = i - r * cols;
    dst[bt * nm + (size_t)c * rows + r] = (bf16)src[bt * nm + i];
  }
}
// lm_weight [32,512] f32 -> [64,512] bf16, rows 32..63 zero (pad to tile width)
__global__ void cvt_pad_lmT_kernel(const float* __restrict__ src, bf16* __restrict__ dst) {
  int i = blockIdx.x * 256 + threadIdx.x;
  if (i < 64 * Dd) {
    int r = i >> 9;
    dst[i] = (r < Vv) ? (bf16)src[i] : (bf16)0.0f;
  }
}

// ------------------------------------------------------------- WMMA GEMM ----
// C[M,N] = act((A_bf16[M,K] @ B, bias) * scale) (+ res);  B given K-major as
// BT[Npad][K].  Block: 256 thr = 8 waves, 64x64 tile; K mult of 32, M mult of 64.
__global__ void __launch_bounds__(256) gemm_kernel(
    const bf16* __restrict__ A, const bf16* __restrict__ BT,
    const float* __restrict__ bias, const float* __restrict__ res,
    float* __restrict__ outF, bf16* __restrict__ outB,
    int M, int N, int K, int act, float scale)
{
  __shared__ __align__(32) bf16 sA[64 * 32];
  int tid = threadIdx.x;
  int wave = tid >> 5, lane = tid & 31, half = lane >> 4;
  int m0 = blockIdx.y * 64, n0 = blockIdx.x * 64;
  int wr = wave >> 1;            // row tile 0..3
  int wc = (wave & 1) * 2;       // col tile base 0 or 2
  v8f acc[2]; acc[0] = vzero8(); acc[1] = vzero8();
  int arow = tid >> 2, acol = (tid & 3) * 8;   // A tile 64x32: one uint4/thread
  const bf16* aptr = A + (size_t)(m0 + arow) * K + acol;
  for (int k0 = 0; k0 < K; k0 += 32) {
    uint4 av = *(const uint4*)(aptr + k0);
    v16bf b0 = load_bt_frag(BT, K, n0 + (wc + 0) * 16, k0);
    v16bf b1 = load_bt_frag(BT, K, n0 + (wc + 1) * 16, k0);
    if (k0 + 32 < K) __builtin_prefetch(aptr + k0 + 32, 0, 1);
    __syncthreads();
    *(uint4*)(sA + arow * 32 + acol) = av;
    __syncthreads();
    v16bf af = load_a_frag(sA, 32, wr * 16, 0);
    acc[0] = wmma_bf16(af, b0, acc[0]);
    acc[1] = wmma_bf16(af, b1, acc[1]);
  }
#pragma unroll
  for (int t = 0; t < 2; ++t) {
    int n = n0 + (wc + t) * 16 + (lane & 15);
    if (n < N) {
      float bv = bias ? bias[n] : 0.0f;
#pragma unroll
      for (int v = 0; v < 8; ++v) {
        int m = m0 + wr * 16 + v + 8 * half;
        float val = (acc[t][v] + bv) * scale;
        if (act == 1) val = gelu_f(val);
        if (res)  val += res[(size_t)m * N + n];
        if (outF) outF[(size_t)m * N + n] = val;
        if (outB) outB[(size_t)m * N + n] = (bf16)val;
      }
    }
  }
}

// ------------------------------------------------- fused GBF pair-bias ------
// Per block: 64 pairs. g[64,128] -> gelu(g@W1+b1)[64,128] -> (@W2+b2)[64,64]
// scattered to input_bias[B,H,L,L] with -inf on padded keys. W1^T/W2^T K-major.
__global__ void __launch_bounds__(256) gbf_kernel(
    const float* __restrict__ dist, const int* __restrict__ edge, const int* __restrict__ tok,
    const float* __restrict__ means, const float* __restrict__ stds,
    const float* __restrict__ mulW, const float* __restrict__ addW,
    const bf16* __restrict__ w1T, const float* __restrict__ b1,
    const bf16* __restrict__ w2T, const float* __restrict__ b2,
    float* __restrict__ outBias)
{
  __shared__ __align__(32) bf16 sWT[128 * 128];  // W1^T, then W2^T
  __shared__ __align__(32) bf16 sT[64 * 128];    // g, then gelu(h1)
  __shared__ float sMean[128], sStd[128], sB1[128], sB2[64], sTval[64];
  int tid = threadIdx.x;
  int base = blockIdx.x * 64;
  if (tid < 128) {
    sMean[tid] = means[tid];
    sStd[tid]  = fabsf(stds[tid]) + 1e-5f;
    sB1[tid]   = b1[tid];
  }
  if (tid < 64) {
    sB2[tid] = b2[tid];
    int p = base + tid;
    int e = edge[p];
    sTval[tid] = mulW[e] * dist[p] + addW[e];
  }
  for (int t = tid; t < (128 * 128) / 8; t += 256)
    ((uint4*)sWT)[t] = ((const uint4*)w1T)[t];
  __syncthreads();
  for (int idx = tid; idx < 64 * 128; idx += 256) {
    int p = idx >> 7, k = idx & 127;
    float x = (sTval[p] - sMean[k]) / sStd[k];
    float g = __expf(-0.5f * x * x) / (2.50662708f * sStd[k]);
    sT[p * 128 + k] = (bf16)g;
  }
  __syncthreads();
  int wave = tid >> 5, lane = tid & 31, half = lane >> 4;
  int wr = wave >> 1;
  int cb = (wave & 1) * 4;                 // 4 of 8 col tiles per wave
  v8f acc[4];
#pragma unroll
  for (int t = 0; t < 4; ++t) acc[t] = vzero8();
#pragma unroll
  for (int k0 = 0; k0 < 128; k0 += 32) {
    v16bf af = load_a_frag(sT, 128, wr * 16, k0);
#pragma unroll
    for (int t = 0; t < 4; ++t)
      acc[t] = wmma_bf16(af, load_bt_frag(sWT, 128, (cb + t) * 16, k0), acc[t]);
  }
  __syncthreads();                          // all reads of sT / sWT done
#pragma unroll
  for (int t = 0; t < 4; ++t) {
    int n = (cb + t) * 16 + (lane & 15);
#pragma unroll
    for (int v = 0; v < 8; ++v) {
      int m = wr * 16 + v + 8 * half;
      sT[m * 128 + n] = (bf16)gelu_f(acc[t][v] + sB1[n]);
    }
  }
  for (int t = tid; t < (64 * 128) / 8; t += 256)   // W2^T [64][128]
    ((uint4*)sWT)[t] = ((const uint4*)w2T)[t];
  __syncthreads();
  int cb2 = (wave & 1) * 2;                 // 2 of 4 col tiles per wave
  v8f o[2]; o[0] = vzero8(); o[1] = vzero8();
#pragma unroll
  for (int k0 = 0; k0 < 128; k0 += 32) {
    v16bf af = load_a_frag(sT, 128, wr * 16, k0);
    o[0] = wmma_bf16(af, load_bt_frag(sWT, 128, (cb2 + 0) * 16, k0), o[0]);
    o[1] = wmma_bf16(af, load_bt_frag(sWT, 128, (cb2 + 1) * 16, k0), o[1]);
  }
#pragma unroll
  for (int t = 0; t < 2; ++t) {
    int h = (cb2 + t) * 16 + (lane & 15);
#pragma unroll
    for (int v = 0; v < 8; ++v) {
      int m = wr * 16 + v + 8 * half;
      int p = base + m;
      int b = p >> 16, i = (p >> 8) & 255, j = p & 255;
      float val = o[t][v] + sB2[h];
      bool padj = (tok[(b << 8) + j] == 0);
      outBias[(((size_t)b * Hh + h) << 16) + ((size_t)i << 8) + j] =
          padj ? -__builtin_inff() : val;
    }
  }
}

// ------------------------------------------------------- elementwise/LN -----
__global__ void embed_ln_kernel(const int* __restrict__ tok, const float* __restrict__ emb,
                                const float* __restrict__ s, const float* __restrict__ b,
                                float* __restrict__ xF, bf16* __restrict__ xB)
{
  int r = blockIdx.x * 256 + threadIdx.x;
  if (r >= BL) return;
  int t = tok[r];
  const float* row = emb + (size_t)t * Dd;
  float m = 0.f;
  for (int c = 0; c < Dd; ++c) m += row[c];
  m *= (1.0f / Dd);
  float var = 0.f;
  for (int c = 0; c < Dd; ++c) { float d = row[c] - m; var += d * d; }
  var *= (1.0f / Dd);
  float rstd = rsqrtf(var + 1e-5f);
  float npad = (t != 0) ? 1.0f : 0.0f;
  for (int c = 0; c < Dd; ++c) {
    float v = ((row[c] - m) * rstd * s[c] + b[c]) * npad;
    xF[(size_t)r * Dd + c] = v;
    xB[(size_t)r * Dd + c] = (bf16)v;
  }
}

__global__ void ln_rows_kernel(const float* __restrict__ in, const float* __restrict__ s,
                               const float* __restrict__ b, float* __restrict__ outF,
                               bf16* __restrict__ outB, int R, int C)
{
  int r = blockIdx.x * 256 + threadIdx.x;
  if (r >= R) return;
  const float* row = in + (size_t)r * C;
  float m = 0.f;
  for (int c = 0; c < C; ++c) m += row[c];
  m /= (float)C;
  float var = 0.f;
  for (int c = 0; c < C; ++c) { float d = row[c] - m; var += d * d; }
  var /= (float)C;
  float rstd = rsqrtf(var + 1e-5f);
  for (int c = 0; c < C; ++c) {
    float v = (row[c] - m) * rstd * s[c] + b[c];
    if (outF) outF[(size_t)r * C + c] = v;
    if (outB) outB[(size_t)r * C + c] = (bf16)v;
  }
}

// ------------------------------------------------------- attention ----------
// One block per (b,h). Online softmax; raw logits stored (next layer's bias).
__global__ void __launch_bounds__(256) attn_kernel(
    const float* __restrict__ q, const float* __restrict__ k, const float* __restrict__ v,
    const float* __restrict__ biasIn, float* __restrict__ biasOut, bf16* __restrict__ avB)
{
  int bh = blockIdx.x, b = bh >> 6, h = bh & 63;
  __shared__ float sk[Ll * 8], sv[Ll * 8];
  int tid = threadIdx.x;
  {
    const float* kp = k + ((size_t)(b * Ll + tid)) * Dd + h * 8;
    const float* vp = v + ((size_t)(b * Ll + tid)) * Dd + h * 8;
#pragma unroll
    for (int d = 0; d < 8; ++d) { sk[tid * 8 + d] = kp[d]; sv[tid * 8 + d] = vp[d]; }
  }
  __syncthreads();
  int i = tid;
  float qr[8];
  const float* qp = q + ((size_t)(b * Ll + i)) * Dd + h * 8;
#pragma unroll
  for (int d = 0; d < 8; ++d) qr[d] = qp[d];
  size_t rowoff = (((size_t)(b * Hh + h)) * Ll + i) * Ll;
  const float* brow = biasIn + rowoff;
  float* orow = biasOut + rowoff;
  float m = -1e30f, s = 0.f, acc[8];
#pragma unroll
  for (int d = 0; d < 8; ++d) acc[d] = 0.f;
  for (int j = 0; j < Ll; ++j) {
    float dot = 0.f;
#pragma unroll
    for (int d = 0; d < 8; ++d) dot += qr[d] * sk[j * 8 + d];
    float lr = dot + brow[j];            // q pre-scaled by hd^-0.5
    orow[j] = lr;
    float lc = fmaxf(lr, -1e30f);
    float mn = fmaxf(m, lc);
    float corr = __expf(m - mn);
    float w = __expf(lc - mn);
    s = s * corr + w;
#pragma unroll
    for (int d = 0; d < 8; ++d) acc[d] = acc[d] * corr + w * sv[j * 8 + d];
    m = mn;
  }
  float inv = 1.0f / s;
  bf16* op = avB + ((size_t)(b * Ll + i)) * Dd + h * 8;
#pragma unroll
  for (int d = 0; d < 8; ++d) op[d] = (bf16)(acc[d] * inv);
}

// ------------------------------------------------------- losses/prep --------
__global__ void prep_kernel(const int* __restrict__ tok, float* cnt,
                            float* out_xnorm, float* out_dnorm)
{
  int tid = threadIdx.x;
  if (tid < Bb) {
    int c = 0;
    for (int l = 0; l < Ll; ++l) c += (tok[tid * Ll + l] != 0);
    cnt[tid] = (float)c;
  }
  if (tid == 4) *out_dnorm = 0.0f;
  if (tid == 5) *out_xnorm = 0.0f;
}

__global__ void xnorm_kernel(const float* __restrict__ x, const int* __restrict__ tok,
                             float* __restrict__ out)
{
  __shared__ float ssum[Bb];
  __shared__ int   scnt[Bb];
  int tid = threadIdx.x;
  if (tid < Bb) { ssum[tid] = 0.f; scnt[tid] = 0; }
  __syncthreads();
  for (int r = tid; r < BL; r += 256) {
    if (tok[r] != 0) {
      const float* row = x + (size_t)r * Dd;
      float ss = 0.f;
      for (int c = 0; c < Dd; ++c) ss += row[c] * row[c];
      float n = sqrtf(ss + 1e-10f);
      float loss = fmaxf(fabsf(n - 22.62741699796952f) - 1.0f, 0.0f);
      atomicAdd(&ssum[r >> 8], loss);
      atomicAdd(&scnt[r >> 8], 1);
    }
  }
  __syncthreads();
  if (tid == 0) {
    float a = 0.f;
    for (int b = 0; b < Bb; ++b) a += ssum[b] / (1e-10f + (float)scnt[b]);
    *out = a * (1.0f / Bb);
  }
}

// -------------------------------------- pair heads (WMMA, 64 pairs/block) ---
// delta = masked(biasF - inBias); delta_norm loss; head-LN; ap = gelu(d@W1+b1)@w2+b2
__global__ void __launch_bounds__(256) delta_p2c_kernel(
    const float* __restrict__ biasF, const float* __restrict__ inBias,
    const int* __restrict__ tok, const float* __restrict__ cnt,
    const float* __restrict__ headS, const float* __restrict__ headB,
    const bf16* __restrict__ w1T, const float* __restrict__ b1,
    const float* __restrict__ w2, const float* __restrict__ b2,
    float* __restrict__ apbuf, float* __restrict__ dnormOut)
{
  __shared__ float sD[64 * 65];                    // f32 delta, padded stride
  __shared__ __align__(32) bf16 sA[64 * 64];       // normalized bf16 A tile
  __shared__ __align__(32) bf16 sWT[64 * 64];      // W1^T
  __shared__ float red[256];
  __shared__ float sMean[64], sRstd[64], sAp[64];
  __shared__ float sHS[64], sHB[64], sb1v[64], sw2v[64];
  __shared__ float sLoss;
  int tid = threadIdx.x;
  int pr = tid & 63, hq = tid >> 6;                // 4 threads per pair
  int base = blockIdx.x * 64;
  int b = base >> 16, i = (base >> 8) & 255, jbase = base & 255;
  for (int t = tid; t < (64 * 64) / 8; t += 256)
    ((uint4*)sWT)[t] = ((const uint4*)w1T)[t];
  if (tid < 64) {
    sHS[tid] = headS[tid]; sHB[tid] = headB[tid];
    sb1v[tid] = b1[tid];   sw2v[tid] = w2[tid];
    sAp[tid] = 0.f;
  }
  if (tid == 0) sLoss = 0.f;
  bool padi = tok[b * Ll + i] == 0;
  bool padj = tok[b * Ll + jbase + pr] == 0;
  size_t biasRow = (((size_t)b * Hh) << 16) + ((size_t)i << 8) + (jbase + pr);
  float psum = 0.f, pss = 0.f;
  for (int hh = 0; hh < 16; ++hh) {
    int h = hq * 16 + hh;
    size_t idx = biasRow + ((size_t)h << 16);
    float dv = padj ? 0.f : (biasF[idx] - inBias[idx]);
    sD[pr * 65 + h] = dv;
    psum += dv; pss += dv * dv;
  }
  red[tid] = psum;
  __syncthreads();
  float sum4 = 0.f;
  if (tid < 64) sum4 = red[tid] + red[tid + 64] + red[tid + 128] + red[tid + 192];
  __syncthreads();
  red[tid] = pss;
  __syncthreads();
  if (tid < 64) {
    float ss = red[tid] + red[tid + 64] + red[tid + 128] + red[tid + 192];
    float mean = sum4 * (1.f / Hh);
    float var = fmaxf(ss * (1.f / Hh) - mean * mean, 0.f);
    sMean[tid] = mean;
    sRstd[tid] = rsqrtf(var + 1e-5f);
    bool pj = tok[b * Ll + jbase + tid] == 0;
    if (!padi && !pj) {
      float n = sqrtf(ss + 1e-10f);
      float loss = fmaxf(fabsf(n - 8.0f) - 1.0f, 0.f);
      float cb = cnt[b];
      loss = loss / (1e-10f + cb * cb) * (1.0f / Bb);
      if (loss != 0.f) atomicAdd(&sLoss, loss);
    }
  }
  __syncthreads();
  if (tid == 0 && sLoss != 0.f) atomicAdd(dnormOut, sLoss);
  for (int hh = 0; hh < 16; ++hh) {               // head-LN -> bf16 A tile
    int h = hq * 16 + hh;
    float val = (sD[pr * 65 + h] - sMean[pr]) * sRstd[pr] * sHS[h] + sHB[h];
    sA[pr * 64 + h] = (bf16)val;
  }
  __syncthreads();
  int wave = tid >> 5, lane = tid & 31, half = lane >> 4;
  int wr = wave >> 1, wc = (wave & 1) * 2;
  v8f acc[2]; acc[0] = vzero8(); acc[1] = vzero8();
#pragma unroll
  for (int k0 = 0; k0 < 64; k0 += 32) {
    v16bf af = load_a_frag(sA, 64, wr * 16, k0);
    acc[0] = wmma_bf16(af, load_bt_frag(sWT, 64, (wc + 0) * 16, k0), acc[0]);
    acc[1] = wmma_bf16(af, load_bt_frag(sWT, 64, (wc + 1) * 16, k0), acc[1]);
  }
#pragma unroll
  for (int t = 0; t < 2; ++t) {
    int n = (wc + t) * 16 + (lane & 15);
    float bn = sb1v[n], wn = sw2v[n];
#pragma unroll
    for (int v = 0; v < 8; ++v) {
      int m = wr * 16 + v + 8 * half;
      atomicAdd(&sAp[m], gelu_f(acc[t][v] + bn) * wn);
    }
  }
  __syncthreads();
  if (tid < 64) apbuf[base + tid] = sAp[tid] + b2[0];
}

// pair_rep = (-inf -> 0)(biasF); h1 = gelu(pr@W+b); LN(h1); d = h1n @ ow + ob
__global__ void __launch_bounds__(256) dist_head_kernel(
    const float* __restrict__ biasF, const bf16* __restrict__ wT,
    const float* __restrict__ bvec, const float* __restrict__ lnS,
    const float* __restrict__ lnB, const float* __restrict__ ow,
    const float* __restrict__ ob, float* __restrict__ dmat)
{
  __shared__ __align__(32) bf16 sA[64 * 64];
  __shared__ __align__(32) bf16 sWT[64 * 64];
  __shared__ float sH[64 * 65];
  __shared__ float sbv[64], sS[64], sB[64], sOW[64];
  int tid = threadIdx.x;
  int pr = tid & 63, hq = tid >> 6;
  int base = blockIdx.x * 64;
  int b = base >> 16, i = (base >> 8) & 255, jbase = base & 255;
  for (int t = tid; t < (64 * 64) / 8; t += 256)
    ((uint4*)sWT)[t] = ((const uint4*)wT)[t];
  if (tid < 64) { sbv[tid] = bvec[tid]; sS[tid] = lnS[tid]; sB[tid] = lnB[tid]; sOW[tid] = ow[tid]; }
  size_t biasRow = (((size_t)b * Hh) << 16) + ((size_t)i << 8) + (jbase + pr);
  for (int hh = 0; hh < 16; ++hh) {
    int h = hq * 16 + hh;
    float vv = biasF[biasRow + ((size_t)h << 16)];
    if (vv == -__builtin_inff()) vv = 0.f;       // isneginf -> 0
    sA[pr * 64 + h] = (bf16)vv;
  }
  __syncthreads();
  int wave = tid >> 5, lane = tid & 31, half = lane >> 4;
  int wr = wave >> 1, wc = (wave & 1) * 2;
  v8f acc[2]; acc[0] = vzero8(); acc[1] = vzero8();
#pragma unroll
  for (int k0 = 0; k0 < 64; k0 += 32) {
    v16bf af = load_a_frag(sA, 64, wr * 16, k0);
    acc[0] = wmma_bf16(af, load_bt_frag(sWT, 64, (wc + 0) * 16, k0), acc[0]);
    acc[1] = wmma_bf16(af, load_bt_frag(sWT, 64, (wc + 1) * 16, k0), acc[1]);
  }
#pragma unroll
  for (int t = 0; t < 2; ++t) {
    int n = (wc + t) * 16 + (lane & 15);
    float bn = sbv[n];
#pragma unroll
    for (int v = 0; v < 8; ++v) {
      int m = wr * 16 + v + 8 * half;
      sH[m * 65 + n] = gelu_f(acc[t][v] + bn);
    }
  }
  __syncthreads();
  if (tid < 64) {
    float sum = 0.f, ss = 0.f;
    for (int o = 0; o < 64; ++o) { float x = sH[tid * 65 + o]; sum += x; ss += x * x; }
    float mean = sum * (1.f / Hh);
    float var = fmaxf(ss * (1.f / Hh) - mean * mean, 0.f);
    float rstd = rsqrtf(var + 1e-5f);
    float dv = ob[0];
    for (int o = 0; o < 64; ++o)
      dv += ((sH[tid * 65 + o] - mean) * rstd * sS[o] + sB[o]) * sOW[o];
    dmat[base + tid] = dv;
  }
}

__global__ void coord_kernel(const float* __restrict__ ap, const float* __restrict__ coord,
                             const int* __restrict__ tok, const float* __restrict__ cnt,
                             float* __restrict__ out)
{
  int idx = blockIdx.x * 256 + threadIdx.x;
  if (idx >= BL) return;
  int b = idx >> 8;
  float ci0 = coord[(size_t)idx * 3], ci1 = coord[(size_t)idx * 3 + 1], ci2 = coord[(size_t)idx * 3 + 2];
  bool padi = tok[idx] == 0;
  float inv = 1.0f / cnt[b];
  float s0 = 0.f, s1 = 0.f, s2 = 0.f;
  const float* aprow = ap + (size_t)idx * Ll;
  for (int j = 0; j < Ll; ++j) {
    if (padi || tok[(b << 8) + j] == 0) continue;
    float w = aprow[j] * inv;
    const float* cj = coord + ((size_t)(b * Ll + j)) * 3;
    s0 += (cj[0] - ci0) * w;
    s1 += (cj[1] - ci1) * w;
    s2 += (cj[2] - ci2) * w;
  }
  out[(size_t)idx * 3]     = ci0 + s0;
  out[(size_t)idx * 3 + 1] = ci1 + s1;
  out[(size_t)idx * 3 + 2] = ci2 + s2;
}

__global__ void sym_kernel(const float* __restrict__ dmat, float* __restrict__ out) {
  size_t p = (size_t)blockIdx.x * 256 + threadIdx.x;
  int b = (int)(p >> 16), i = (int)((p >> 8) & 255), j = (int)(p & 255);
  out[p] = 0.5f * (dmat[p] + dmat[((size_t)b << 16) + ((size_t)j << 8) + i]);
}

// --------------------------------------------------------------- launch -----
static inline void gemm(const bf16* A, const bf16* BT, const float* bias, const float* res,
                        float* outF, bf16* outB, int M, int N, int Npad, int K,
                        int act, float scale, hipStream_t s) {
  dim3 g(Npad / 64, M / 64);
  gemm_kernel<<<g, 256, 0, s>>>(A, BT, bias, res, outF, outB, M, N, K, act, scale);
}
static inline void cvtT(const float* src, bf16* dst, int rows, int cols, int batch,
                        hipStream_t s) {
  dim3 g((rows * cols + 255) / 256, batch);
  cvt_t_kernel<<<g, 256, 0, s>>>(src, dst, rows, cols);
}

extern "C" void kernel_launch(void* const* d_in, const int* in_sizes, int n_in,
                              void* d_out, int out_size, void* d_ws, size_t ws_size,
                              hipStream_t stream)
{
  (void)in_sizes; (void)n_in; (void)out_size; (void)ws_size;
  const int*   tok    = (const int*)  d_in[0];
  const float* dist   = (const float*)d_in[1];
  const float* coord  = (const float*)d_in[2];
  const int*   edge   = (const int*)  d_in[3];
  const float* embedW = (const float*)d_in[4];
  const float* emb_s  = (const float*)d_in[5];
  const float* emb_b  = (const float*)d_in[6];
  const float* wq = (const float*)d_in[7];   const float* bq = (const float*)d_in[8];
  const float* wk = (const float*)d_in[9];   const float* bk = (const float*)d_in[10];
  const float* wv = (const float*)d_in[11];  const float* bv = (const float*)d_in[12];
  const float* wo = (const float*)d_in[13];  const float* bo = (const float*)d_in[14];
  const float* ln1s = (const float*)d_in[15]; const float* ln1b = (const float*)d_in[16];
  const float* w1 = (const float*)d_in[17];  const float* b1 = (const float*)d_in[18];
  const float* w2 = (const float*)d_in[19];  const float* b2 = (const float*)d_in[20];
  const float* ln2s = (const float*)d_in[21]; const float* ln2b = (const float*)d_in[22];
  const float* fin_s = (const float*)d_in[23]; const float* fin_b = (const float*)d_in[24];
  const float* head_s = (const float*)d_in[25]; const float* head_b = (const float*)d_in[26];
  const float* gmeans = (const float*)d_in[27]; const float* gstds = (const float*)d_in[28];
  const float* gmul = (const float*)d_in[29]; const float* gbias = (const float*)d_in[30];
  const float* gw1 = (const float*)d_in[31]; const float* gb1 = (const float*)d_in[32];
  const float* gw2 = (const float*)d_in[33]; const float* gb2 = (const float*)d_in[34];
  const float* lmw = (const float*)d_in[35]; const float* lmb = (const float*)d_in[36];
  const float* lmlns = (const float*)d_in[37]; const float* lmlnb = (const float*)d_in[38];
  const float* lmweight = (const float*)d_in[39]; const float* lmbias = (const float*)d_in[40];
  const float* p2cw1 = (const float*)d_in[41]; const float* p2cb1 = (const float*)d_in[42];
  const float* p2cw2 = (const float*)d_in[43]; const float* p2cb2 = (const float*)d_in[44];
  const float* dhw = (const float*)d_in[45]; const float* dhb = (const float*)d_in[46];
  const float* dhlns = (const float*)d_in[47]; const float* dhlnb = (const float*)d_in[48];
  const float* dhow = (const float*)d_in[49]; const float* dhob = (const float*)d_in[50];

  // ---- workspace bump allocator (256B aligned) ----
  char* ws = (char*)d_ws;
  size_t off = 0;
  auto alloc = [&](size_t bytes) { char* p = ws + off; off += (bytes + 255) & ~(size_t)255; return p; };
  bf16* wqT = (bf16*)alloc((size_t)NLAYERS * Dd * Dd * 2);     // per-layer W^T [out][in]
  bf16* wkT = (bf16*)alloc((size_t)NLAYERS * Dd * Dd * 2);
  bf16* wvT = (bf16*)alloc((size_t)NLAYERS * Dd * Dd * 2);
  bf16* woT = (bf16*)alloc((size_t)NLAYERS * Dd * Dd * 2);
  bf16* w1T = (bf16*)alloc((size_t)NLAYERS * Dd * FFNf * 2);   // [FFN][D] per layer
  bf16* w2T = (bf16*)alloc((size_t)NLAYERS * FFNf * Dd * 2);   // [D][FFN] per layer
  bf16* gw1T = (bf16*)alloc((size_t)Kk * Kk * 2);
  bf16* gw2T = (bf16*)alloc((size_t)Hh * Kk * 2);
  bf16* lmwT = (bf16*)alloc((size_t)Dd * Dd * 2);
  bf16* lmTB = (bf16*)alloc((size_t)64 * Dd * 2);              // lm_weight padded [64][512]
  bf16* p2cw1T = (bf16*)alloc((size_t)Hh * Hh * 2);
  bf16* dhwT   = (bf16*)alloc((size_t)Hh * Hh * 2);
  bf16* xB   = (bf16*)alloc((size_t)BL * Dd * 2);
  bf16* avB  = (bf16*)alloc((size_t)BL * Dd * 2);
  bf16* hB   = (bf16*)alloc((size_t)BL * FFNf * 2);
  bf16* lmhB = (bf16*)alloc((size_t)BL * Dd * 2);
  float* xF  = (float*)alloc((size_t)BL * Dd * 4);
  float* tF  = (float*)alloc((size_t)BL * Dd * 4);
  float* qF  = (float*)alloc((size_t)BL * Dd * 4);
  float* kF  = (float*)alloc((size_t)BL * Dd * 4);
  float* vF  = (float*)alloc((size_t)BL * Dd * 4);
  float* lm1F = (float*)alloc((size_t)BL * Dd * 4);
  float* inBias = (float*)alloc((size_t)BHLL * 4);
  float* biasA  = (float*)alloc((size_t)BHLL * 4);
  float* biasB  = (float*)alloc((size_t)BHLL * 4);
  float* apF    = (float*)alloc((size_t)BLL * 4);
  float* dmatF  = (float*)alloc((size_t)BLL * 4);
  float* cntF   = (float*)alloc(Bb * 4);

  float* outLM    = (float*)d_out;            // [B,L,V]   32768
  float* outDIST  = outLM + 32768;            // [B,L,L]   262144
  float* outCOORD = outDIST + 262144;         // [B,L,3]   3072
  float* outXN    = outCOORD + 3072;          // scalar
  float* outDN    = outXN + 1;                // scalar

  // ---- weight conversion: bf16, K-major (transposed) ----
  cvtT(wq, wqT, Dd, Dd, NLAYERS, stream);
  cvtT(wk, wkT, Dd, Dd, NLAYERS, stream);
  cvtT(wv, wvT, Dd, Dd, NLAYERS, stream);
  cvtT(wo, woT, Dd, Dd, NLAYERS, stream);
  cvtT(w1, w1T, Dd, FFNf, NLAYERS, stream);
  cvtT(w2, w2T, FFNf, Dd, NLAYERS, stream);
  cvtT(gw1, gw1T, Kk, Kk, 1, stream);
  cvtT(gw2, gw2T, Kk, Hh, 1, stream);
  cvtT(lmw, lmwT, Dd, Dd, 1, stream);
  cvt_pad_lmT_kernel<<<(64 * Dd + 255) / 256, 256, 0, stream>>>(lmweight, lmTB);
  cvtT(p2cw1, p2cw1T, Hh, Hh, 1, stream);
  cvtT(dhw, dhwT, Hh, Hh, 1, stream);

  prep_kernel<<<1, 256, 0, stream>>>(tok, cntF, outXN, outDN);

  // ---- GBF pair bias (fused WMMA) -> input_bias ----
  gbf_kernel<<<BLL / 64, 256, 0, stream>>>(dist, edge, tok, gmeans, gstds, gmul, gbias,
                                           gw1T, gb1, gw2T, gb2, inBias);

  // ---- embedding + LN ----
  embed_ln_kernel<<<BL / 256, 256, 0, stream>>>(tok, embedW, emb_s, emb_b, xF, xB);

  const float scaling = 0.35355339059327373f;  // hd^-0.5
  for (int l = 0; l < NLAYERS; ++l) {
    const bf16* WQ = wqT + (size_t)l * Dd * Dd;
    const bf16* WK = wkT + (size_t)l * Dd * Dd;
    const bf16* WV = wvT + (size_t)l * Dd * Dd;
    const bf16* WO = woT + (size_t)l * Dd * Dd;
    const bf16* W1 = w1T + (size_t)l * Dd * FFNf;
    const bf16* W2 = w2T + (size_t)l * FFNf * Dd;
    float* bin  = (l == 0) ? inBias : ((l & 1) ? biasA : biasB);
    float* bout = (l & 1) ? biasB : biasA;

    gemm(xB, WQ, bq + l * Dd, nullptr, qF, nullptr, BL, Dd, Dd, Dd, 0, scaling, stream);
    gemm(xB, WK, bk + l * Dd, nullptr, kF, nullptr, BL, Dd, Dd, Dd, 0, 1.0f, stream);
    gemm(xB, WV, bv + l * Dd, nullptr, vF, nullptr, BL, Dd, Dd, Dd, 0, 1.0f, stream);
    attn_kernel<<<Bb * Hh, 256, 0, stream>>>(qF, kF, vF, bin, bout, avB);
    gemm(avB, WO, bo + l * Dd, xF, tF, nullptr, BL, Dd, Dd, Dd, 0, 1.0f, stream);
    ln_rows_kernel<<<BL / 256, 256, 0, stream>>>(tF, ln1s + l * Dd, ln1b + l * Dd, xF, xB, BL, Dd);
    gemm(xB, W1, b1 + l * FFNf, nullptr, nullptr, hB, BL, FFNf, FFNf, Dd, 1, 1.0f, stream);
    gemm(hB, W2, b2 + l * Dd, xF, tF, nullptr, BL, Dd, Dd, FFNf, 0, 1.0f, stream);
    ln_rows_kernel<<<BL / 256, 256, 0, stream>>>(tF, ln2s + l * Dd, ln2b + l * Dd, xF, xB, BL, Dd);
  }
  float* biasF = biasB;  // layer 7 wrote biasB

  xnorm_kernel<<<1, 256, 0, stream>>>(xF, tok, outXN);
  ln_rows_kernel<<<BL / 256, 256, 0, stream>>>(xF, fin_s, fin_b, nullptr, xB, BL, Dd);

  // ---- LM head ----
  gemm(xB, lmwT, lmb, nullptr, lm1F, nullptr, BL, Dd, Dd, Dd, 1, 1.0f, stream);
  ln_rows_kernel<<<BL / 256, 256, 0, stream>>>(lm1F, lmlns, lmlnb, nullptr, lmhB, BL, Dd);
  gemm(lmhB, lmTB, lmbias, nullptr, outLM, nullptr, BL, Vv, 64, Dd, 0, 1.0f, stream);

  // ---- pair heads (WMMA) ----
  delta_p2c_kernel<<<BLL / 64, 256, 0, stream>>>(biasF, inBias, tok, cntF, head_s, head_b,
                                                 p2cw1T, p2cb1, p2cw2, p2cb2, apF, outDN);
  dist_head_kernel<<<BLL / 64, 256, 0, stream>>>(biasF, dhwT, dhb, dhlns, dhlnb, dhow, dhob, dmatF);
  coord_kernel<<<BL / 256, 256, 0, stream>>>(apF, coord, tok, cntF, outCOORD);
  sym_kernel<<<BLL / 256, 256, 0, stream>>>(dmatF, outDIST);
}
